// MS_SSIM_58884001628835
// MI455X (gfx1250) — matrix operand order repaired and checked by reading, not compiled, not run on hardware
//
#include <hip/hip_runtime.h>
#include <hip/hip_bf16.h>

typedef __attribute__((ext_vector_type(16))) _Float16 v16h;
typedef __attribute__((ext_vector_type(8)))  _Float16 v8h;
typedef __attribute__((ext_vector_type(4)))  _Float16 v4h;
typedef __attribute__((ext_vector_type(8)))  float    v8f;
typedef __attribute__((ext_vector_type(4)))  float    v4f;

#define PLANES 48            // 16 batch * 3 channels
#define C1_CONST 1.0e-4f     // (0.01*L)^2, L = 1
#define C2_CONST 9.0e-4f     // (0.03*L)^2, L = 1

// Gaussian weight g[d], d in [0,10], sigma = 1.5  (normalized by invS)
__device__ __forceinline__ float gw(int d, float invS) {
    float dd = (float)(d - 5);
    return __expf(-dd * dd / 4.5f) * invS;
}

// ---------------------------------------------------------------------------
// Init: zero accumulators and precompute the banded Gaussian weight matrices
// in per-lane WMMA operand layout (Bw: 32x16 B-layout, Aw: 16x32 A-layout).
// One v16h (32 B) per lane per matrix; level kernels just load them.
// ---------------------------------------------------------------------------
__global__ __launch_bounds__(32)
void msssim_init_kernel(float* __restrict__ accum,
                        v16h* __restrict__ BwTab, v16h* __restrict__ AwTab)
{
    const int lane = threadIdx.x;
    if (lane < 16) accum[lane] = 0.f;

    float S = 0.f;
#pragma unroll
    for (int k = 0; k < 11; ++k) { float dd = (float)(k - 5); S += __expf(-dd * dd / 4.5f); }
    const float invS = 1.f / S;

    // B-layout (32x16): element (K,N)=g[K-N]; lane -> N=lane&15, K=(h<8?h:h+8)+Koff, Koff=(lane<16?0:8)
    v16h Bw;
    {
        int N = lane & 15, Koff = (lane < 16) ? 0 : 8;
#pragma unroll
        for (int h = 0; h < 16; ++h) {
            int d = Koff + ((h < 8) ? h : h + 8) - N;
            Bw[h] = (_Float16)((d >= 0 && d <= 10) ? gw(d, invS) : 0.f);
        }
    }
    // A-layout (16x32): element (M,K)=g[K-M]; lane -> M=lane&15;
    // half h -> K = h + (lane<16 ? 0 : 16)
    v16h Aw;
    {
        int M = lane & 15, koff = (lane < 16) ? 0 : 16;
#pragma unroll
        for (int h = 0; h < 16; ++h) {
            int K = h + koff;
            int d = K - M;
            Aw[h] = (_Float16)((d >= 0 && d <= 10) ? gw(d, invS) : 0.f);
        }
    }
    BwTab[lane] = Bw;
    AwTab[lane] = Aw;
}

// ---------------------------------------------------------------------------
// One wave32 per 16x16 tile of the VALID conv output.
// Grid: (tilesX, tilesX, PLANES) -> no integer division in the kernel.
// ---------------------------------------------------------------------------
__global__ __launch_bounds__(32)
void msssim_level_kernel(const float* __restrict__ x, const float* __restrict__ y,
                         int H, int W,
                         float* __restrict__ accum,
                         const v16h* __restrict__ BwTab, const v16h* __restrict__ AwTab)
{
    const int lane  = threadIdx.x;          // 0..31 (wave32)
    const int tc    = blockIdx.x;
    const int tr    = blockIdx.y;
    const int plane = blockIdx.z;

    const int outH = H - 10, outW = W - 10;
    const int r0 = tr * 16, c0 = tc * 16;
    const float* __restrict__ xp = x + (size_t)plane * H * W;
    const float* __restrict__ yp = y + (size_t)plane * H * W;

    // Precomputed banded Gaussian operands (4x global_load_b128, L2-resident)
    const v16h Bw = BwTab[lane];
    const v16h Aw = AwTab[lane];

    // f16 input patches, row stride 48 halves (96 B) -> 16B-aligned rows, staggered banks
    __shared__ __align__(16) _Float16 Xs[32][48];
    __shared__ __align__(16) _Float16 Ys[32][48];
    // Horizontal-pass intermediate, column-major f16: T[field][col][row]
    __shared__ __align__(32) _Float16 T[5][16][32];

    // ---- Stage 32x32 f32 patch -> f16 LDS, fully coalesced b128 loads.
    // W and c0 are multiples of 16 at every level, chunks are 4-wide, so each
    // chunk is either fully valid or feeds only masked outputs.
    {
        int sub   = lane >> 3;              // 0..3  (row within group of 4)
        int chunk = lane & 7;               // 0..7  (4-float column chunk)
        int ccRel = chunk * 4;
        int cc    = c0 + ccRel;
        int ccl   = (cc + 3 <= W - 1) ? cc : 0;   // OOB chunk -> safe address
#pragma unroll
        for (int i = 0; i < 8; ++i) {
            int rloc = i * 4 + sub;
            int rr = r0 + rloc; if (rr > H - 1) rr = H - 1;
            v4f xv = *(const v4f*)(xp + (size_t)rr * W + ccl);
            v4f yv = *(const v4f*)(yp + (size_t)rr * W + ccl);
            v4h xh4, yh4;
#pragma unroll
            for (int j = 0; j < 4; ++j) { xh4[j] = (_Float16)xv[j]; yh4[j] = (_Float16)yv[j]; }
            *(v4h*)&Xs[rloc][ccRel] = xh4;
            *(v4h*)&Ys[rloc][ccRel] = yh4;
        }
    }
    __syncthreads();

    // ---- Gather A-layout operands from LDS: two aligned ds_load_b128 per tile.
    // A-layout per lane: M = lane&15; halves 0..7 -> K=koff..koff+7,
    // halves 8..15 -> K=16+koff..16+koff+7, koff = (lane<16 ? 0 : 8).
    v16h xh[2], yh[2];
    {
        int Mrow = lane & 15;
        int koff = (lane < 16) ? 0 : 8;
#pragma unroll
        for (int t = 0; t < 2; ++t) {
            int r = t * 16 + Mrow;
            v8h xlo = *(const v8h*)&Xs[r][koff];
            v8h xhi = *(const v8h*)&Xs[r][16 + koff];
            v8h ylo = *(const v8h*)&Ys[r][koff];
            v8h yhi = *(const v8h*)&Ys[r][16 + koff];
            xh[t] = __builtin_shufflevector(xlo, xhi, 0,1,2,3,4,5,6,7,8,9,10,11,12,13,14,15);
            yh[t] = __builtin_shufflevector(ylo, yhi, 0,1,2,3,4,5,6,7,8,9,10,11,12,13,14,15);
        }
    }

    // ---- Horizontal pass: D(16x16) = In(16x32) * Bw(32x16), 2 row-tiles x 5 fields
#pragma unroll
    for (int f = 0; f < 5; ++f) {
#pragma unroll
        for (int t = 0; t < 2; ++t) {
            v16h a;
            if      (f == 0) a = xh[t];
            else if (f == 1) a = yh[t];
            else if (f == 2) a = xh[t] * xh[t];
            else if (f == 3) a = yh[t] * yh[t];
            else             a = xh[t] * yh[t];
            v8f c = {};
            v8f d = __builtin_amdgcn_wmma_f32_16x16x32_f16(
                        false, a, false, Bw, (short)0, c, false, false);
            // D layout: VGPR r holds M=r (lanes 0-15) / M=r+8 (lanes 16-31), N=lane&15
            int col   = lane & 15;
            int rbase = t * 16 + ((lane < 16) ? 0 : 8);
#pragma unroll
            for (int r = 0; r < 8; ++r)
                T[f][col][rbase + r] = (_Float16)d[r];
        }
    }
    __syncthreads();

    // ---- Vertical pass: D(16x16) = Aw(16x32) * T(32x16)
    // B layout per lane: N = lane&15, K = h + (lane<16 ? 0 : 16) -> 16 contiguous rows
    v8f res[5];
    {
        int colB  = lane & 15;
        int KoffB = (lane < 16) ? 0 : 16;
#pragma unroll
        for (int f = 0; f < 5; ++f) {
            v16h b = *(const v16h*)(&T[f][colB][KoffB]);
            v8f c = {};
            res[f] = __builtin_amdgcn_wmma_f32_16x16x32_f16(
                         false, Aw, false, b, (short)0, c, false, false);
        }
    }

    // ---- Pointwise SSIM / CS with edge masking, then wave reduction
    float ssum = 0.f, csum = 0.f;
    {
        int Mbase = (lane < 16) ? 0 : 8;
        int colg  = c0 + (lane & 15);
#pragma unroll
        for (int r = 0; r < 8; ++r) {
            float mu1 = res[0][r], mu2 = res[1][r];
            float exx = res[2][r], eyy = res[3][r], exy = res[4][r];
            float mu1s = mu1 * mu1, mu2s = mu2 * mu2, mu12 = mu1 * mu2;
            float s1 = exx - mu1s, s2 = eyy - mu2s, s12 = exy - mu12;
            float v1 = 2.f * s12 + C2_CONST;
            float v2 = s1 + s2 + C2_CONST;
            float cs = v1 / v2;
            float sm = ((2.f * mu12 + C1_CONST) * v1) / ((mu1s + mu2s + C1_CONST) * v2);
            int rowg = r0 + Mbase + r;
            bool ok = (rowg < outH) && (colg < outW);
            ssum += ok ? sm : 0.f;
            csum += ok ? cs : 0.f;
        }
    }
#pragma unroll
    for (int off = 16; off > 0; off >>= 1) {
        ssum += __shfl_xor(ssum, off, 32);   // butterfly: all lanes end with total
        csum += __shfl_xor(csum, off, 32);
    }
    if (lane == 0)      atomicAdd(&accum[0], ssum);
    else if (lane == 1) atomicAdd(&accum[1], csum);
}

// ---------------------------------------------------------------------------
// Fused 2x2 average pool for both images (NCHW, per-plane).
// Grid: (ceil(Wo/256), Ho, planes) -> no integer division.
// ---------------------------------------------------------------------------
__global__ __launch_bounds__(256)
void msssim_pool2_kernel(const float* __restrict__ inx, const float* __restrict__ iny,
                         float* __restrict__ outx, float* __restrict__ outy,
                         int Ho, int Wo)
{
    int c = blockIdx.x * blockDim.x + threadIdx.x;
    if (c >= Wo) return;
    int r = blockIdx.y;
    int p = blockIdx.z;
    size_t hw  = (size_t)Ho * Wo;
    size_t oidx = (size_t)p * hw + (size_t)r * Wo + c;
    int Wi = 2 * Wo;
    size_t ibase = (size_t)p * 4 * hw + (size_t)(2 * r) * Wi + 2 * c;

    float2 x0 = *(const float2*)(inx + ibase);
    float2 x1 = *(const float2*)(inx + ibase + Wi);
    float2 y0 = *(const float2*)(iny + ibase);
    float2 y1 = *(const float2*)(iny + ibase + Wi);
    outx[oidx] = 0.25f * (x0.x + x0.y + x1.x + x1.y);
    outy[oidx] = 0.25f * (y0.x + y0.y + y1.x + y1.y);
}

// Final combine: replicates prod(pow1[:-1] * pow2[-1]) exactly
// = (prod_{l<4} mcs[l]^w[l]) * (mssim[4]^w[4])^4
__global__ void msssim_finish_kernel(const float* __restrict__ accum, float* __restrict__ out)
{
    const float w[5] = {0.0448f, 0.2856f, 0.3001f, 0.2363f, 0.1333f};
    float res = 1.f;
#pragma unroll
    for (int l = 0; l < 5; ++l) {
        int Hl = 512 >> l;
        float cnt = (float)PLANES * (float)(Hl - 10) * (float)(Hl - 10);
        float ms = (accum[2 * l]     / cnt + 1.f) * 0.5f;
        float mc = (accum[2 * l + 1] / cnt + 1.f) * 0.5f;
        // values are strictly positive -> exp/log pow
        if (l < 4) res *= __expf(w[l] * __logf(mc));
        else       res *= __expf(4.f * w[4] * __logf(ms));
    }
    out[0] = res;
}

extern "C" void kernel_launch(void* const* d_in, const int* in_sizes, int n_in,
                              void* d_out, int out_size, void* d_ws, size_t ws_size,
                              hipStream_t stream)
{
    (void)in_sizes; (void)n_in; (void)out_size; (void)ws_size;
    const float* x = (const float*)d_in[0];
    const float* y = (const float*)d_in[1];
    float* out = (float*)d_out;
    char*  ws  = (char*)d_ws;

    float* accum = (float*)ws;                    // 16 floats
    v16h*  BwTab = (v16h*)(ws + 128);             // 32 * 32 B
    v16h*  AwTab = (v16h*)(ws + 128 + 1024);      // 32 * 32 B
    size_t off = 4096;                            // pyramid buffers start here

    msssim_init_kernel<<<dim3(1), dim3(32), 0, stream>>>(accum, BwTab, AwTab);

    for (int l = 0; l < 5; ++l) {
        int Hl = 512 >> l;
        int outH = Hl - 10;
        int tiles = (outH + 15) / 16;
        dim3 grid((unsigned)tiles, (unsigned)tiles, PLANES);
        msssim_level_kernel<<<grid, dim3(32), 0, stream>>>(x, y, Hl, Hl,
                                                           accum + 2 * l, BwTab, AwTab);
        if (l < 4) {
            int Ho = Hl / 2;
            size_t planeBytes = (size_t)PLANES * Ho * Ho * sizeof(float);
            float* nx = (float*)(ws + off); off += planeBytes;
            float* ny = (float*)(ws + off); off += planeBytes;
            dim3 pgrid((unsigned)((Ho + 255) / 256), (unsigned)Ho, PLANES);
            msssim_pool2_kernel<<<pgrid, dim3(256), 0, stream>>>(x, y, nx, ny, Ho, Ho);
            x = nx; y = ny;
        }
    }

    msssim_finish_kernel<<<dim3(1), dim3(1), 0, stream>>>(accum, out);
}
